// BatchedGraphSAGEMean1_40862318854441
// MI455X (gfx1250) — matrix operand order repaired and checked by reading, not compiled, not run on hardware
//
#include <hip/hip_runtime.h>

// ---------------------------------------------------------------------------
// BatchedGraphSAGE (mean aggregation) for MI455X / gfx1250, wave32 + WMMA.
//   B=8, N=2048, K=32, F_IN=F_OUT=256, channels out = 512.
// Memory-bound problem (~90MB HBM, gather is L2-resident) -> use bf16 WMMA
// (v_wmma_f32_16x16x32_bf16) for the two projections, f32 accumulate, and
// fuse L2-normalize + ReLU + BN partial stats into the GEMM epilogue.
// ---------------------------------------------------------------------------

typedef __attribute__((ext_vector_type(16))) __bf16 v16bf;
typedef __attribute__((ext_vector_type(8)))  float  v8f;

struct alignas(16) BF8 { __bf16 h[8]; };   // one 16-byte bf16 chunk

constexpr int BB   = 8;
constexpr int NN   = 2048;
constexpr int KNB  = 32;          // neighbors
constexpr int F    = 256;         // F_IN == F_OUT
constexpr int CO   = 512;         // 2*F_OUT concat channels
constexpr int ROWS = BB * NN;     // 16384 (b,n) rows
constexpr int TILE_R = 32;        // rows per GEMM workgroup
constexpr int GEMM_BLOCKS = ROWS / TILE_R;   // 512

// ---------------------------------------------------------------- converts
__global__ void k_f2bf(const float* __restrict__ s, __bf16* __restrict__ d, int n) {
  int i = blockIdx.x * blockDim.x + threadIdx.x;
  if (i < n) d[i] = (__bf16)s[i];
}

// -------------------------------------------------------------- gather-mean
// One block per (b,n) row; thread = feature. idx is uniform per block ->
// scalar loads; x rows are L2-resident (16.8MB << 192MB L2).
__global__ void k_gather_mean(const float* __restrict__ x,
                              const int* __restrict__ idx,
                              __bf16* __restrict__ nbuf) {
  const int bx = blockIdx.x;                 // 0..16383
  const int b  = bx >> 11;                   // / N
  const int n  = bx & (NN - 1);
  const int f  = threadIdx.x;                // 0..255
  const float* xb = x + (size_t)b * NN * F;
  float acc = 0.f;
#pragma unroll 4
  for (int k = 0; k < KNB; ++k) {
    const int j = idx[n * KNB + k];
    acc += xb[(size_t)j * F + f];
  }
  nbuf[(size_t)bx * F + f] = (__bf16)(acc * (1.0f / KNB));
}

// ------------------------------------------------------------------- GEMM
// 8 waves: waves 0-3 -> h_self (cols 0..255) from xb, waves 4-7 -> h_nei
// (cols 256..511) from nbuf. Each wave: 32 rows x 64 cols = 2x4 WMMA tiles.
// Epilogue: +bias, per-row L2 norm (cross-wave via LDS, deterministic),
// ReLU, store, per-channel (sum, sum^2) partials per block.
__global__ __launch_bounds__(256) void k_gemm(
    const __bf16* __restrict__ xb,  const __bf16* __restrict__ nbuf,
    const __bf16* __restrict__ wxb, const __bf16* __restrict__ wnb,
    const float* __restrict__ biasx, const float* __restrict__ biasn,
    float* __restrict__ out, float* __restrict__ partials) {

  __shared__ float s_rowsqW[TILE_R][8];   // per-row, per-wave partial sum-of-squares
  __shared__ float s_rowsq[TILE_R];       // final per-row sum-of-squares
  __shared__ float s_colstat[8][64][2];   // per-wave column (sum, sum^2)

  const int tid  = threadIdx.x;
  const int wave = tid >> 5;
  const int lane = tid & 31;
  const int l16  = lane & 15;
  const bool hi  = lane >= 16;
  const int half = wave >> 2;              // 0 = self, 1 = neighbor
  const int cw   = wave & 3;
  const int colW = cw * 64;                // col base inside the 256-wide half
  const int row_base = blockIdx.x * TILE_R;

  const __bf16* A  = half ? nbuf : xb;
  const __bf16* W  = half ? wnb  : wxb;
  const float* bias = half ? biasn : biasx;

  v8f acc[2][4] = {};

  // bf16 A-fragment chunking per ISA 7.12.2: lanes 0-15 hold K {0..7,16..23},
  // lanes 16-31 hold K {8..15,24..31}. B fragment: lane = column, same K
  // chunking as A (lanes 0-15: K {0..7,16..23}, lanes 16-31: K {8..15,24..31}).
  const int koA = hi ? 8 : 0;
  const int koB = hi ? 8 : 0;

#pragma unroll 2
  for (int k0 = 0; k0 < F; k0 += 32) {
    v16bf a[2], b[4];
#pragma unroll
    for (int rt = 0; rt < 2; ++rt) {
      const __bf16* p = A + (size_t)(row_base + rt * 16 + l16) * F + k0 + koA;
      if (k0 + 32 < F) __builtin_prefetch(p + 32, 0, 0);   // global_prefetch_b8
      const BF8 c0 = *reinterpret_cast<const BF8*>(p);
      const BF8 c1 = *reinterpret_cast<const BF8*>(p + 16);
#pragma unroll
      for (int i = 0; i < 8; ++i) { a[rt][i] = c0.h[i]; a[rt][i + 8] = c1.h[i]; }
    }
#pragma unroll
    for (int ct = 0; ct < 4; ++ct) {
      const __bf16* p = W + (size_t)(colW + ct * 16 + l16) * F + k0 + koB;
      const BF8 c0 = *reinterpret_cast<const BF8*>(p);
      const BF8 c1 = *reinterpret_cast<const BF8*>(p + 16);
#pragma unroll
      for (int i = 0; i < 8; ++i) { b[ct][i] = c0.h[i]; b[ct][i + 8] = c1.h[i]; }
    }
#pragma unroll
    for (int rt = 0; rt < 2; ++rt)
#pragma unroll
      for (int ct = 0; ct < 4; ++ct)
        acc[rt][ct] = __builtin_amdgcn_wmma_f32_16x16x32_bf16(
            false, a[rt], false, b[ct], (short)0, acc[rt][ct], false, false);
  }

  // ---- bias + per-row sum of squares (pre-normalize values) ----
  const int rbase = hi ? 8 : 0;
#pragma unroll
  for (int rt = 0; rt < 2; ++rt) {
    float rs[8] = {0.f, 0.f, 0.f, 0.f, 0.f, 0.f, 0.f, 0.f};
#pragma unroll
    for (int ct = 0; ct < 4; ++ct) {
      const float bv = bias[colW + ct * 16 + l16];
#pragma unroll
      for (int i = 0; i < 8; ++i) {
        const float h = acc[rt][ct][i] + bv;
        acc[rt][ct][i] = h;
        rs[i] += h * h;
      }
    }
    // reduce across the 16 lanes that share each row (different columns)
#pragma unroll
    for (int i = 0; i < 8; ++i) {
      float v = rs[i];
      v += __shfl_xor(v, 1, 16);
      v += __shfl_xor(v, 2, 16);
      v += __shfl_xor(v, 4, 16);
      v += __shfl_xor(v, 8, 16);
      if (l16 == 0) s_rowsqW[rt * 16 + rbase + i][wave] = v;   // plain store
    }
  }
  __syncthreads();
  if (tid < TILE_R) {                       // deterministic cross-wave sum
    float s = 0.f;
#pragma unroll
    for (int w = 0; w < 8; ++w) s += s_rowsqW[tid][w];
    s_rowsq[tid] = s;
  }
  __syncthreads();

  float rscale[2][8];
#pragma unroll
  for (int rt = 0; rt < 2; ++rt)
#pragma unroll
    for (int i = 0; i < 8; ++i) {
      const float s = s_rowsq[rt * 16 + rbase + i];
      rscale[rt][i] = 1.0f / fmaxf(sqrtf(s), 1e-12f);
    }

  // ---- normalize + ReLU + store + per-channel partial stats ----
  float cs[4][2] = {};
#pragma unroll
  for (int rt = 0; rt < 2; ++rt)
#pragma unroll
    for (int ct = 0; ct < 4; ++ct) {
      const int col = wave * 64 + ct * 16 + l16;      // global channel
#pragma unroll
      for (int i = 0; i < 8; ++i) {
        const int rl = rt * 16 + rbase + i;
        float h = fmaxf(acc[rt][ct][i] * rscale[rt][i], 0.f);
        out[(size_t)(row_base + rl) * CO + col] = h;
        cs[ct][0] += h;
        cs[ct][1] += h * h;
      }
    }
#pragma unroll
  for (int ct = 0; ct < 4; ++ct) {
    const float a0 = cs[ct][0] + __shfl_xor(cs[ct][0], 16, 32);
    const float a1 = cs[ct][1] + __shfl_xor(cs[ct][1], 16, 32);
    if (!hi) { s_colstat[wave][ct * 16 + l16][0] = a0;
               s_colstat[wave][ct * 16 + l16][1] = a1; }
  }
  __syncthreads();

  float* pb = partials + (size_t)blockIdx.x * (CO * 2);
  for (int i = tid; i < CO * 2; i += 256) {
    const int c = i >> 1;
    pb[i] = s_colstat[c >> 6][c & 63][i & 1];
  }
}

// ------------------------------------------------------------- BN reduce
__global__ void k_bn_reduce(const float* __restrict__ partials,
                            const float* __restrict__ gamma,
                            const float* __restrict__ beta,
                            float* __restrict__ ss) {
  const int c = blockIdx.x * blockDim.x + threadIdx.x;
  if (c >= CO) return;
  float s0 = 0.f, s1 = 0.f;
  for (int b = 0; b < GEMM_BLOCKS; ++b) {   // fixed order -> deterministic
    s0 += partials[(size_t)b * (CO * 2) + c * 2];
    s1 += partials[(size_t)b * (CO * 2) + c * 2 + 1];
  }
  const float invN = 1.0f / (float)ROWS;
  const float mu  = s0 * invN;
  const float var = fmaxf(s1 * invN - mu * mu, 0.f);
  const float sc  = gamma[c] * rsqrtf(var + 1e-5f);
  ss[c] = sc;
  ss[CO + c] = beta[c] - mu * sc;
}

// ------------------------------------------------------------- BN apply
__global__ void k_bn_apply(float* __restrict__ out, const float* __restrict__ ss) {
  const size_t i = ((size_t)blockIdx.x * blockDim.x + threadIdx.x) * 4;
  if (i >= (size_t)ROWS * CO) return;
  float4 v = *reinterpret_cast<float4*>(out + i);
  const int c = (int)(i & (CO - 1));
  v.x = v.x * ss[c]     + ss[CO + c];
  v.y = v.y * ss[c + 1] + ss[CO + c + 1];
  v.z = v.z * ss[c + 2] + ss[CO + c + 2];
  v.w = v.w * ss[c + 3] + ss[CO + c + 3];
  *reinterpret_cast<float4*>(out + i) = v;
}

// ---------------------------------------------------------------- launch
extern "C" void kernel_launch(void* const* d_in, const int* in_sizes, int n_in,
                              void* d_out, int out_size, void* d_ws, size_t ws_size,
                              hipStream_t stream) {
  const float* x      = (const float*)d_in[0];   // [8,2048,256]
  const int*   idx    = (const int*)  d_in[1];   // [2048,32]
  const float* Wx_w   = (const float*)d_in[2];   // [256,256]
  const float* Wx_b   = (const float*)d_in[3];   // [256]
  const float* Wn_w   = (const float*)d_in[4];   // [256,256]
  const float* Wn_b   = (const float*)d_in[5];   // [256]
  const float* gamma  = (const float*)d_in[6];   // [512]
  const float* beta   = (const float*)d_in[7];   // [512]
  float* out = (float*)d_out;                    // [8,2048,512]

  // workspace layout (bytes), all 16B aligned
  char* ws = (char*)d_ws;
  __bf16* xb    = (__bf16*)(ws);                         //  8,388,608
  __bf16* nbuf  = (__bf16*)(ws + 8388608);               //  8,388,608
  __bf16* wxb   = (__bf16*)(ws + 16777216);              //    131,072
  __bf16* wnb   = (__bf16*)(ws + 16908288);              //    131,072
  float*  part  = (float*)(ws + 17039360);               //  2,097,152
  float*  ss    = (float*)(ws + 19136512);               //      4,096

  // 1) bf16 conversions (x and weights)
  k_f2bf<<<(ROWS * F) / 256, 256, 0, stream>>>(x, xb, ROWS * F);
  k_f2bf<<<(F * F) / 256,    256, 0, stream>>>(Wx_w, wxb, F * F);
  k_f2bf<<<(F * F) / 256,    256, 0, stream>>>(Wn_w, wnb, F * F);
  // 2) neighbor gather-mean (L2-resident)
  k_gather_mean<<<ROWS, 256, 0, stream>>>(x, idx, nbuf);
  // 3) fused dual-projection WMMA GEMM + L2-norm + ReLU + BN partials
  k_gemm<<<GEMM_BLOCKS, 256, 0, stream>>>(xb, nbuf, wxb, wnb, Wx_b, Wn_b,
                                          out, part);
  // 4) BN statistics reduction -> scale/shift
  k_bn_reduce<<<2, 256, 0, stream>>>(part, gamma, beta, ss);
  // 5) BN affine apply, vectorized in place
  k_bn_apply<<<(ROWS * CO) / (4 * 256), 256, 0, stream>>>(out, ss);
}
